// transformer_classification_46961172415149
// MI455X (gfx1250) — compile-verified, matrix operand order checked by reading, not run on hardware
//
#include <hip/hip_runtime.h>

// ---------------------------------------------------------------------------
// ViT encoder (4 shared layers) + head, bf16 WMMA (v_wmma_f32_16x16x32_bf16)
// B=64, T=197 tokens, D=768, H=12, DH=64, HID=3072, classes=1000
// GEMMs: one wave32 computes a 32x64 tile (8 WMMA accumulators) to double
// weight (B-operand) reuse vs a 16x64 tile -> ~21 flop/byte from L2.
// ---------------------------------------------------------------------------

typedef __attribute__((ext_vector_type(16))) __bf16 v16bf;
typedef __attribute__((ext_vector_type(8)))  float  v8f;

#define MTOK   12608      // 64*197  (= 32 * 394)
#define DMODEL 768
#define NHEAD  12
#define DHEAD  64
#define HID    3072
#define SPAD   224        // 197 padded to 7*32
#define SEQ    197
#define NB     64

struct Frag {
    union { v16bf v; uint4 q[2]; };
};

// A/B operand fragment: lane (lo,hi) holds 8 bf16 at [base .. base+7] and
// 8 bf16 at [base+16 .. base+23] where base already includes hi*8.
__device__ __forceinline__ Frag load_frag(const unsigned short* p) {
    Frag f;
    f.q[0] = *(const uint4*)(p);
    f.q[1] = *(const uint4*)(p + 16);
    return f;
}

__device__ __forceinline__ v8f wmma_bf16(const Frag& a, const Frag& b, v8f c) {
    return __builtin_amdgcn_wmma_f32_16x16x32_bf16(
        false, a.v, false, b.v, (short)0, c, false, false);
}

__device__ __forceinline__ unsigned short f2bf(float f) {
    unsigned int u = __float_as_uint(f);
    unsigned int r = u + 0x7fffu + ((u >> 16) & 1u);   // round-to-nearest-even
    return (unsigned short)(r >> 16);
}

// ---------------------------------------------------------------------------
// Generic bf16 GEMM: D[M,N] = A[M,K] * Bt[N,K]^T (+bias)(+relu)(+resid)
// One wave per 32x64 tile: 2 row-subtiles x 4 col-subtiles = 8 WMMA / k-step.
// ---------------------------------------------------------------------------
__global__ __launch_bounds__(32) void gemm_bias_kernel(
    const unsigned short* __restrict__ A, const unsigned short* __restrict__ Bt,
    const float* __restrict__ bias, const float* __restrict__ resid,
    float* __restrict__ outF, unsigned short* __restrict__ outB,
    int M, int N, int K, int relu)
{
    const int lane = threadIdx.x;
    const int lo = lane & 15, hi = lane >> 4;
    const int mbase = blockIdx.x * 32;
    const int nbase = blockIdx.y * 64;

    v8f acc[2][4] = {};
    const unsigned short* arow0 = A  + (size_t)(mbase + lo) * K + (hi << 3);
    const unsigned short* arow1 = arow0 + (size_t)16 * K;
    const unsigned short* brow  = Bt + (size_t)(nbase + lo) * K + (hi << 3);
    const size_t bstep = (size_t)16 * K;

    for (int kk = 0; kk < K; kk += 32) {
        Frag a0 = load_frag(arow0 + kk);
        Frag a1 = load_frag(arow1 + kk);
#pragma unroll
        for (int d = 0; d < 4; ++d) {
            Frag b = load_frag(brow + (size_t)d * bstep + kk);
            acc[0][d] = wmma_bf16(a0, b, acc[0][d]);
            acc[1][d] = wmma_bf16(a1, b, acc[1][d]);
        }
    }

#pragma unroll
    for (int d = 0; d < 4; ++d) {
        int n = nbase + d * 16 + lo;
        if (n >= N) continue;
        float bv = bias ? bias[n] : 0.0f;
#pragma unroll
        for (int e = 0; e < 2; ++e) {
#pragma unroll
            for (int r = 0; r < 8; ++r) {
                int row = mbase + e * 16 + r + (hi << 3);
                float v = acc[e][d][r] + bv;
                if (relu) v = v > 0.0f ? v : 0.0f;
                size_t o = (size_t)row * N + n;
                if (resid) v += resid[o];
                if (outF) outF[o] = v;
                if (outB) outB[o] = f2bf(v);
            }
        }
    }
}

// ---------------------------------------------------------------------------
// QKV GEMM: grid.z selects q/k/v. 32x64 tile per wave. Epilogue scatters
// into per-head layouts:
//   q,k : [B][H][SPAD][64] bf16      v : [B][H][64][SPAD] bf16 (transposed)
// ---------------------------------------------------------------------------
__global__ __launch_bounds__(32) void gemm_qkv_kernel(
    const unsigned short* __restrict__ A,
    const unsigned short* __restrict__ wqT, const unsigned short* __restrict__ wkT,
    const unsigned short* __restrict__ wvT,
    unsigned short* __restrict__ qh, unsigned short* __restrict__ kh,
    unsigned short* __restrict__ vT)
{
    const int lane = threadIdx.x;
    const int lo = lane & 15, hi = lane >> 4;
    const int mbase = blockIdx.x * 32;
    const int h = blockIdx.y;                 // head == 64-col group
    const int z = blockIdx.z;
    const unsigned short* Bt = (z == 0) ? wqT : (z == 1) ? wkT : wvT;
    const int K = DMODEL;

    v8f acc[2][4] = {};
    const unsigned short* arow0 = A  + (size_t)(mbase + lo) * K + (hi << 3);
    const unsigned short* arow1 = arow0 + (size_t)16 * K;
    const unsigned short* brow  = Bt + (size_t)(h * 64 + lo) * K + (hi << 3);

    for (int kk = 0; kk < K; kk += 32) {
        Frag a0 = load_frag(arow0 + kk);
        Frag a1 = load_frag(arow1 + kk);
#pragma unroll
        for (int d = 0; d < 4; ++d) {
            Frag b = load_frag(brow + (size_t)(d * 16) * K + kk);
            acc[0][d] = wmma_bf16(a0, b, acc[0][d]);
            acc[1][d] = wmma_bf16(a1, b, acc[1][d]);
        }
    }

#pragma unroll
    for (int e = 0; e < 2; ++e) {
#pragma unroll
        for (int r = 0; r < 8; ++r) {
            int m = mbase + e * 16 + r + (hi << 3);
            int b = m / SEQ, s = m - b * SEQ;
            size_t bh = (size_t)b * NHEAD + h;
#pragma unroll
            for (int d = 0; d < 4; ++d) {
                int dh = d * 16 + lo;
                unsigned short v = f2bf(acc[e][d][r]);
                if (z == 0)      qh[(bh * SPAD + s) * DHEAD + dh] = v;
                else if (z == 1) kh[(bh * SPAD + s) * DHEAD + dh] = v;
                else             vT[(bh * DHEAD + dh) * SPAD + s] = v;
            }
        }
    }
}

// ---------------------------------------------------------------------------
// Flash attention: one wave per (qtile, head, batch). Online softmax,
// scores and P*V both on WMMA. Writes outA = x + attn (fused residual).
// ---------------------------------------------------------------------------
__global__ __launch_bounds__(32) void attention_kernel(
    const unsigned short* __restrict__ qh, const unsigned short* __restrict__ kh,
    const unsigned short* __restrict__ vT, const float* __restrict__ x,
    float* __restrict__ outA)
{
    __shared__ __align__(16) unsigned short ldsP[16 * 32];

    const int lane = threadIdx.x;
    const int lo = lane & 15, hi = lane >> 4;
    const int qt = blockIdx.x, h = blockIdx.y, b = blockIdx.z;
    const size_t bh = (size_t)b * NHEAD + h;

    const unsigned short* qbase = qh + bh * SPAD * DHEAD;
    const unsigned short* kbase = kh + bh * SPAD * DHEAD;
    const unsigned short* vbase = vT + bh * DHEAD * SPAD;

    Frag qa0 = load_frag(qbase + (size_t)(qt * 16 + lo) * DHEAD + (hi << 3));
    Frag qa1 = load_frag(qbase + (size_t)(qt * 16 + lo) * DHEAD + 32 + (hi << 3));

    v8f o[4] = {};
    float mrow[8], lrow[8];
#pragma unroll
    for (int r = 0; r < 8; ++r) { mrow[r] = -1e30f; lrow[r] = 0.0f; }

    for (int j = 0; j < SPAD / 32; ++j) {
        const int kb = j * 32;
        // ---- scores for two 16-key subtiles (K = DHEAD = 64, two k-steps)
        v8f s0 = {}, s1 = {};
        {
            Frag k0a = load_frag(kbase + (size_t)(kb + lo) * DHEAD + (hi << 3));
            Frag k0b = load_frag(kbase + (size_t)(kb + lo) * DHEAD + 32 + (hi << 3));
            s0 = wmma_bf16(qa0, k0a, s0);
            s0 = wmma_bf16(qa1, k0b, s0);
            Frag k1a = load_frag(kbase + (size_t)(kb + 16 + lo) * DHEAD + (hi << 3));
            Frag k1b = load_frag(kbase + (size_t)(kb + 16 + lo) * DHEAD + 32 + (hi << 3));
            s1 = wmma_bf16(qa0, k1a, s1);
            s1 = wmma_bf16(qa1, k1b, s1);
        }
        const float scale = 0.125f;                     // 1/sqrt(64)
        const bool mask0 = (kb + lo)      >= SEQ;
        const bool mask1 = (kb + 16 + lo) >= SEQ;

        float p0[8], p1[8];
#pragma unroll
        for (int r = 0; r < 8; ++r) {
            float a0 = mask0 ? -1e30f : s0[r] * scale;
            float a1 = mask1 ? -1e30f : s1[r] * scale;
            float vmax = fmaxf(a0, a1);                 // row r+8*hi, col = lo
            vmax = fmaxf(vmax, __shfl_xor(vmax, 1));
            vmax = fmaxf(vmax, __shfl_xor(vmax, 2));
            vmax = fmaxf(vmax, __shfl_xor(vmax, 4));
            vmax = fmaxf(vmax, __shfl_xor(vmax, 8));
            float mnew = fmaxf(mrow[r], vmax);
            float cold = __expf(mrow[r] - mnew);
            float e0 = __expf(a0 - mnew);
            float e1 = __expf(a1 - mnew);
            float rs = e0 + e1;
            rs += __shfl_xor(rs, 1);
            rs += __shfl_xor(rs, 2);
            rs += __shfl_xor(rs, 4);
            rs += __shfl_xor(rs, 8);
            lrow[r] = lrow[r] * cold + rs;
            mrow[r] = mnew;
            p0[r] = e0; p1[r] = e1;
#pragma unroll
            for (int d = 0; d < 4; ++d) o[d][r] = o[d][r] * cold;
        }

        // ---- reshape P (C-layout) -> A-layout via LDS, as bf16 [16 x 32]
#pragma unroll
        for (int r = 0; r < 8; ++r) {
            int qr = r + (hi << 3);
            ldsP[qr * 32 + lo]      = f2bf(p0[r]);
            ldsP[qr * 32 + 16 + lo] = f2bf(p1[r]);
        }
        __syncthreads();
        Frag pa;
        {
            const unsigned short* pp = ldsP + lo * 32 + (hi << 3);
            pa.q[0] = *(const uint4*)(pp);
            pa.q[1] = *(const uint4*)(pp + 16);
        }
        __syncthreads();

        // ---- o += P (16x32) * V (32x16) per dh tile
#pragma unroll
        for (int d = 0; d < 4; ++d) {
            Frag vf = load_frag(vbase + (size_t)(d * 16 + lo) * SPAD + kb + (hi << 3));
            o[d] = wmma_bf16(pa, vf, o[d]);
        }
    }

    // ---- normalize and store (fused residual: outA = x + attn)
    const int scol = h * DHEAD;
#pragma unroll
    for (int r = 0; r < 8; ++r) {
        int s = qt * 16 + r + (hi << 3);
        if (s >= SEQ) continue;
        float inv = 1.0f / lrow[r];
        size_t rowg = ((size_t)b * SEQ + s) * DMODEL;
#pragma unroll
        for (int d = 0; d < 4; ++d) {
            int col = scol + d * 16 + lo;
            outA[rowg + col] = x[rowg + col] + o[d][r] * inv;
        }
    }
}

// ---------------------------------------------------------------------------
// LayerNorm (f32 in) -> bf16 out. One 256-thread block per token row.
// ---------------------------------------------------------------------------
__global__ __launch_bounds__(256) void layernorm_bf16_kernel(
    const float* __restrict__ x, const float* __restrict__ g,
    const float* __restrict__ be, unsigned short* __restrict__ out)
{
    __shared__ float red[256];
    const int row = blockIdx.x, t = threadIdx.x;
    const float* xr = x + (size_t)row * DMODEL;
    float v0 = xr[t], v1 = xr[t + 256], v2 = xr[t + 512];

    red[t] = v0 + v1 + v2; __syncthreads();
    for (int o = 128; o > 0; o >>= 1) { if (t < o) red[t] += red[t + o]; __syncthreads(); }
    float mu = red[0] * (1.0f / DMODEL);
    __syncthreads();
    float d0 = v0 - mu, d1 = v1 - mu, d2 = v2 - mu;
    red[t] = d0 * d0 + d1 * d1 + d2 * d2; __syncthreads();
    for (int o = 128; o > 0; o >>= 1) { if (t < o) red[t] += red[t + o]; __syncthreads(); }
    float rstd = rsqrtf(red[0] * (1.0f / DMODEL) + 1e-5f);

    unsigned short* orow = out + (size_t)row * DMODEL;
    orow[t]       = f2bf(d0 * rstd * g[t]       + be[t]);
    orow[t + 256] = f2bf(d1 * rstd * g[t + 256] + be[t + 256]);
    orow[t + 512] = f2bf(d2 * rstd * g[t + 512] + be[t + 512]);
}

// ---------------------------------------------------------------------------
// x[b, s, :] = (s==0 ? cls : image[b, s-1]) + pos[s]
// ---------------------------------------------------------------------------
__global__ void embed_kernel(const float* __restrict__ image,
                             const float* __restrict__ cls,
                             const float* __restrict__ pos,
                             float* __restrict__ x)
{
    int idx = blockIdx.x * 256 + threadIdx.x;
    if (idx >= MTOK * DMODEL) return;
    int m = idx / DMODEL, d = idx - m * DMODEL;
    int b = m / SEQ, s = m - b * SEQ;
    float v = (s == 0) ? cls[d] : image[((size_t)b * (SEQ - 1) + (s - 1)) * DMODEL + d];
    x[idx] = v + pos[(size_t)s * DMODEL + d];
}

// weight f32 [K,N] -> bf16 transposed [Npad,K] (zero-padded rows)
__global__ void transpose_bf16_kernel(const float* __restrict__ src,
                                      unsigned short* __restrict__ dst,
                                      int K, int N, int Npad)
{
    int idx = blockIdx.x * 256 + threadIdx.x;
    if (idx >= Npad * K) return;
    int n = idx / K, k = idx - n * K;
    dst[idx] = (n < N) ? f2bf(src[(size_t)k * N + n]) : (unsigned short)0;
}

__global__ void cls_extract_kernel(const float* __restrict__ x,
                                   unsigned short* __restrict__ out)
{
    int idx = blockIdx.x * 256 + threadIdx.x;
    if (idx >= NB * DMODEL) return;
    int b = idx / DMODEL, d = idx - b * DMODEL;
    out[idx] = f2bf(x[((size_t)b * SEQ) * DMODEL + d]);
}

// ---------------------------------------------------------------------------
extern "C" void kernel_launch(void* const* d_in, const int* in_sizes, int n_in,
                              void* d_out, int out_size, void* d_ws, size_t ws_size,
                              hipStream_t stream)
{
    (void)in_sizes; (void)n_in; (void)out_size; (void)ws_size;
    const float* image  = (const float*)d_in[0];
    const float* clstok = (const float*)d_in[1];
    const float* pos    = (const float*)d_in[2];
    const float* ln1_g  = (const float*)d_in[3];
    const float* ln1_b  = (const float*)d_in[4];
    const float* wq     = (const float*)d_in[5];
    const float* wk     = (const float*)d_in[6];
    const float* wv     = (const float*)d_in[7];
    const float* ln2_g  = (const float*)d_in[8];
    const float* ln2_b  = (const float*)d_in[9];
    const float* w1     = (const float*)d_in[10];
    const float* b1     = (const float*)d_in[11];
    const float* w2     = (const float*)d_in[12];
    const float* b2     = (const float*)d_in[13];
    const float* head_w = (const float*)d_in[14];
    const float* head_b = (const float*)d_in[15];

    char* ws = (char*)d_ws;
    size_t off = 0;
    auto alloc = [&](size_t bytes) -> char* {
        char* p = ws + off;
        off = (off + bytes + 255) & ~(size_t)255;
        return p;
    };
    const size_t headBytes = (size_t)NB * NHEAD * SPAD * DHEAD * 2;  // 22 MB each

    float*          bufX = (float*)alloc((size_t)MTOK * DMODEL * 4);
    float*          bufA = (float*)alloc((size_t)MTOK * DMODEL * 4);
    unsigned short* xnB  = (unsigned short*)alloc((size_t)MTOK * DMODEL * 2);
    unsigned short* hidB = (unsigned short*)alloc((size_t)MTOK * HID * 2);
    unsigned short* qhB  = (unsigned short*)alloc(headBytes);
    unsigned short* khB  = (unsigned short*)alloc(headBytes);
    unsigned short* vTB  = (unsigned short*)alloc(headBytes);
    unsigned short* wqT  = (unsigned short*)alloc((size_t)DMODEL * DMODEL * 2);
    unsigned short* wkT  = (unsigned short*)alloc((size_t)DMODEL * DMODEL * 2);
    unsigned short* wvT  = (unsigned short*)alloc((size_t)DMODEL * DMODEL * 2);
    unsigned short* w1T  = (unsigned short*)alloc((size_t)HID * DMODEL * 2);
    unsigned short* w2T  = (unsigned short*)alloc((size_t)DMODEL * HID * 2);
    unsigned short* hwT  = (unsigned short*)alloc((size_t)1024 * DMODEL * 2);
    unsigned short* clsB = (unsigned short*)alloc((size_t)NB * DMODEL * 2);

    // zero per-head buffers once so padding rows (s >= 197) are clean
    hipMemsetAsync(qhB, 0, headBytes, stream);
    hipMemsetAsync(khB, 0, headBytes, stream);
    hipMemsetAsync(vTB, 0, headBytes, stream);

    auto nblk = [](size_t n) { return (unsigned)((n + 255) / 256); };

    transpose_bf16_kernel<<<nblk((size_t)768 * 768),  256, 0, stream>>>(wq, wqT, 768, 768, 768);
    transpose_bf16_kernel<<<nblk((size_t)768 * 768),  256, 0, stream>>>(wk, wkT, 768, 768, 768);
    transpose_bf16_kernel<<<nblk((size_t)768 * 768),  256, 0, stream>>>(wv, wvT, 768, 768, 768);
    transpose_bf16_kernel<<<nblk((size_t)3072 * 768), 256, 0, stream>>>(w1, w1T, 768, 3072, 3072);
    transpose_bf16_kernel<<<nblk((size_t)768 * 3072), 256, 0, stream>>>(w2, w2T, 3072, 768, 768);
    transpose_bf16_kernel<<<nblk((size_t)1024 * 768), 256, 0, stream>>>(head_w, hwT, 768, 1000, 1024);

    embed_kernel<<<nblk((size_t)MTOK * DMODEL), 256, 0, stream>>>(image, clstok, pos, bufX);

    for (int layer = 0; layer < 4; ++layer) {
        layernorm_bf16_kernel<<<MTOK, 256, 0, stream>>>(bufX, ln1_g, ln1_b, xnB);
        gemm_qkv_kernel<<<dim3(MTOK / 32, NHEAD, 3), 32, 0, stream>>>(
            xnB, wqT, wkT, wvT, qhB, khB, vTB);
        attention_kernel<<<dim3(13, NHEAD, NB), 32, 0, stream>>>(
            qhB, khB, vTB, bufX, bufA);
        layernorm_bf16_kernel<<<MTOK, 256, 0, stream>>>(bufA, ln2_g, ln2_b, xnB);
        gemm_bias_kernel<<<dim3(MTOK / 32, HID / 64), 32, 0, stream>>>(
            xnB, w1T, b1, nullptr, nullptr, hidB, MTOK, HID, DMODEL, 1);
        gemm_bias_kernel<<<dim3(MTOK / 32, DMODEL / 64), 32, 0, stream>>>(
            hidB, w2T, b2, bufA, bufX, nullptr, MTOK, DMODEL, HID, 0);
    }

    cls_extract_kernel<<<nblk((size_t)NB * DMODEL), 256, 0, stream>>>(bufX, clsB);
    gemm_bias_kernel<<<dim3(NB / 32, 1024 / 64), 32, 0, stream>>>(
        clsB, hwT, head_b, nullptr, (float*)d_out, nullptr, NB, 1000, DMODEL, 0);
}
